// MemModule_31602369364491
// MI455X (gfx1250) — matrix operand order repaired and checked by reading, not compiled, not run on hardware
//
#include <hip/hip_runtime.h>

typedef __attribute__((ext_vector_type(16))) _Float16 v16h;
typedef __attribute__((ext_vector_type(8)))  float    v8f;
typedef __attribute__((ext_vector_type(4)))  float    v4f;

#define NROWS  65536
#define FEA    512
#define MEMD   2000
#define MEMP   2016      // padded to multiple of 32
#define SSTR   2020      // padded f32 LDS row stride (bank-conflict free)
#define LAMBDA 0.0025f
#define EPSV   1e-12f

// ---------- prep: W f32 -> Wh f16 [MEMP][FEA] (row major) + Wth f16 [FEA][MEMP] (transposed)
__global__ void prep_weights(const float* __restrict__ W,
                             _Float16* __restrict__ Wh,
                             _Float16* __restrict__ Wth)
{
    int idx = blockIdx.x * 256 + threadIdx.x;
    if (idx >= MEMP * FEA) return;
    int m = idx / FEA;
    int f = idx - m * FEA;
    float v = (m < MEMD) ? W[m * FEA + f] : 0.0f;
    _Float16 h = (_Float16)v;
    Wh[m * FEA + f]   = h;
    Wth[f * MEMP + m] = h;
}

// Convert two 8-float chunks (p0..p0+7 and p0+16..p0+23) into one f16 A/B fragment.
__device__ __forceinline__ v16h cvt_frag(const float* p0)
{
    v4f c0 = *(const v4f*)(p0);
    v4f c1 = *(const v4f*)(p0 + 4);
    v4f c2 = *(const v4f*)(p0 + 16);
    v4f c3 = *(const v4f*)(p0 + 20);
    v16h av;
    av[0]  = (_Float16)c0[0]; av[1]  = (_Float16)c0[1]; av[2]  = (_Float16)c0[2]; av[3]  = (_Float16)c0[3];
    av[4]  = (_Float16)c1[0]; av[5]  = (_Float16)c1[1]; av[6]  = (_Float16)c1[2]; av[7]  = (_Float16)c1[3];
    av[8]  = (_Float16)c2[0]; av[9]  = (_Float16)c2[1]; av[10] = (_Float16)c2[2]; av[11] = (_Float16)c2[3];
    av[12] = (_Float16)c3[0]; av[13] = (_Float16)c3[1]; av[14] = (_Float16)c3[2]; av[15] = (_Float16)c3[3];
    return av;
}

// ---------- fused: GEMM1 (x@W^T) -> softmax/shrink/renorm -> GEMM2 (att@W)
__launch_bounds__(256)
__global__ void memae_fused(const float* __restrict__ x,
                            const _Float16* __restrict__ Wh,
                            const _Float16* __restrict__ Wth,
                            float* __restrict__ outO,
                            float* __restrict__ outA)
{
    extern __shared__ float S[];              // [16][SSTR] f32 scores/att
    const int tid  = threadIdx.x;
    const int wave = tid >> 5;
    const int lane = tid & 31;
    const int ln   = lane & 15;               // M (GEMM A row) / N (B,C,D column)
    const int g    = lane >> 4;               // lane half-group
    const int row0 = blockIdx.x << 4;         // 16 x-rows per workgroup

    // ---- stage full A (16 rows x K=512) in registers as f16 fragments
    v16h a[16];
    {
        const float* xr = x + (size_t)(row0 + ln) * FEA;
        #pragma unroll
        for (int ks = 0; ks < 16; ++ks)
            a[ks] = cvt_frag(xr + ks * 32 + 8 * g);
    }

    // ---- GEMM1: each wave sweeps m-tiles t, t+8, ...
    for (int t = wave; t < MEMP / 16; t += 8) {
        const int m0 = t << 4;
        v8f acc = {0.f,0.f,0.f,0.f,0.f,0.f,0.f,0.f};
        const _Float16* wr = Wh + (size_t)(m0 + ln) * FEA + 8 * g;
        #pragma unroll
        for (int ks = 0; ks < 16; ++ks) {
            union { v16h h; uint4 u[2]; } b;
            b.u[0] = *(const uint4*)(wr + ks * 32);
            b.u[1] = *(const uint4*)(wr + ks * 32 + 16);
            acc = __builtin_amdgcn_wmma_f32_16x16x32_f16(false, a[ks], false, b.h,
                                                         (short)0, acc, false, false);
        }
        #pragma unroll
        for (int r = 0; r < 8; ++r)
            S[(r + 8 * g) * SSTR + m0 + ln] = acc[r];
    }
    __syncthreads();

    // ---- softmax + hard-shrink-relu + L1 renorm; 16 lanes per row (2000 = 125*16, no divergence)
    {
        const int r = tid >> 4;
        const int c = tid & 15;
        float* Sr = S + r * SSTR;

        float mx = -3.402823466e38f;
        for (int m = c; m < MEMD; m += 16) mx = fmaxf(mx, Sr[m]);
        #pragma unroll
        for (int off = 8; off; off >>= 1) mx = fmaxf(mx, __shfl_xor(mx, off, 16));

        float se = 0.0f;
        for (int m = c; m < MEMD; m += 16) {
            float e = __expf(Sr[m] - mx);
            Sr[m] = e;
            se += e;
        }
        #pragma unroll
        for (int off = 8; off; off >>= 1) se += __shfl_xor(se, off, 16);
        float invZ = 1.0f / se;

        float sa = 0.0f;
        for (int m = c; m < MEMD; m += 16) {
            float p = Sr[m] * invZ;
            float d = p - LAMBDA;
            float av = (d > 0.0f) ? (d * p) / (d + EPSV) : 0.0f;   // relu(p-l)*p/(|p-l|+eps)
            Sr[m] = av;
            sa += av;
        }
        #pragma unroll
        for (int off = 8; off; off >>= 1) sa += __shfl_xor(sa, off, 16);
        float sc = 1.0f / fmaxf(sa, EPSV);

        float* ar = outA + (size_t)(row0 + r) * MEMD;
        for (int m = c; m < MEMD; m += 16) {
            float v = Sr[m] * sc;
            Sr[m] = v;
            ar[m] = v;
        }
        Sr[MEMD + c] = 0.0f;                  // zero pad columns 2000..2015
    }
    __syncthreads();

    // ---- GEMM2: out = att @ W ; each wave owns 4 N-tiles (64 f-columns)
    v8f oacc[4];
    #pragma unroll
    for (int i = 0; i < 4; ++i) oacc[i] = (v8f){0.f,0.f,0.f,0.f,0.f,0.f,0.f,0.f};
    const int f0 = wave * 64;
    const float* Sa = S + ln * SSTR;
    for (int ks = 0; ks < MEMP / 32; ++ks) {
        v16h av = cvt_frag(Sa + ks * 32 + 8 * g);    // A from LDS att, shared by 4 tiles
        #pragma unroll
        for (int nt = 0; nt < 4; ++nt) {
            const _Float16* wr = Wth + (size_t)(f0 + nt * 16 + ln) * MEMP + ks * 32 + 8 * g;
            union { v16h h; uint4 u[2]; } b;
            b.u[0] = *(const uint4*)(wr);
            b.u[1] = *(const uint4*)(wr + 16);
            oacc[nt] = __builtin_amdgcn_wmma_f32_16x16x32_f16(false, av, false, b.h,
                                                              (short)0, oacc[nt], false, false);
        }
    }
    #pragma unroll
    for (int nt = 0; nt < 4; ++nt) {
        const int fc = f0 + nt * 16 + ln;
        #pragma unroll
        for (int r = 0; r < 8; ++r)
            outO[(size_t)(row0 + r + 8 * g) * FEA + fc] = oacc[nt][r];
    }
}

extern "C" void kernel_launch(void* const* d_in, const int* in_sizes, int n_in,
                              void* d_out, int out_size, void* d_ws, size_t ws_size,
                              hipStream_t stream)
{
    (void)in_sizes; (void)n_in; (void)out_size; (void)ws_size;
    const float* x = (const float*)d_in[0];
    const float* W = (const float*)d_in[1];
    float* outO = (float*)d_out;
    float* outA = outO + (size_t)NROWS * FEA;

    _Float16* Wh  = (_Float16*)d_ws;                    // MEMP*FEA f16  (~2.06 MB)
    _Float16* Wth = Wh + (size_t)MEMP * FEA;            // FEA*MEMP f16  (~2.06 MB)

    const int prepN = MEMP * FEA;
    prep_weights<<<(prepN + 255) / 256, 256, 0, stream>>>(W, Wh, Wth);

    const size_t lds_bytes = (size_t)16 * SSTR * sizeof(float);   // 129,280 B
    memae_fused<<<NROWS / 16, 256, lds_bytes, stream>>>(x, Wh, Wth, outO, outA);
}
